// Generator_75350906241749
// MI455X (gfx1250) — matrix operand order repaired and checked by reading, not compiled, not run on hardware
//
#include <hip/hip_runtime.h>
#include <hip/hip_bf16.h>

typedef __attribute__((ext_vector_type(16))) _Float16 v16h;
typedef __attribute__((ext_vector_type(8)))  float    v8f;
typedef unsigned int u32x4 __attribute__((ext_vector_type(4)));
typedef int          i32x8 __attribute__((ext_vector_type(8)));
typedef int          i32x4 __attribute__((ext_vector_type(4)));

#define NEG_SLOPE 0.2f
#define NNODES (256 * 196)   // 50176
#define KDIM 64

// ---------- monotonic float <-> uint encoding (for atomicMax on floats) ----
__device__ __forceinline__ unsigned f2mono(float f) {
    unsigned u = __float_as_uint(f);
    return (u & 0x80000000u) ? ~u : (u | 0x80000000u);
}
__device__ __forceinline__ float mono2f(unsigned u) {
    return (u & 0x80000000u) ? __uint_as_float(u ^ 0x80000000u)
                             : __uint_as_float(~u);
}

// ---------- A fragment (16x32 f16): lane holds K = k0+half*16+[0..15] -------
__device__ __forceinline__ v16h load_a_frag(const float* __restrict__ ap) {
    // ap already points at A[row*K + k0 + half*16]
    v16h a;
#pragma unroll
    for (int i = 0; i < 8; ++i) a[i] = (_Float16)ap[i];
#pragma unroll
    for (int i = 0; i < 8; ++i) a[8 + i] = (_Float16)ap[8 + i];
    return a;
}

// ---------- TDM: DMA a 2-D f32 tile (rows x cols, row_stride) into LDS ------
#if defined(__gfx1250__) && __has_builtin(__builtin_amdgcn_tensor_load_to_lds)
#define HAVE_TDM 1
__device__ __forceinline__ void tdm_load_2d_f32(unsigned lds_addr,
                                                const float* gptr,
                                                int rows, int cols,
                                                int row_stride) {
    unsigned long long ga = (unsigned long long)(uintptr_t)gptr;
    u32x4 g0;
    g0[0] = 1u;                                   // count=1 valid descriptor
    g0[1] = lds_addr;                             // LDS byte address
    g0[2] = (unsigned)ga;                         // global_addr[31:0]
    g0[3] = (unsigned)((ga >> 32) & 0x01FFFFFFull) | (2u << 30); // [56:32]|type=2
    i32x8 g1;
    g1[0] = (2 << 16);                            // data_size = 4 bytes
    g1[1] = (int)((unsigned)(cols & 0xFFFF) << 16);        // tensor_dim0 lo16
    g1[2] = (int)(((unsigned)cols >> 16) |
                  ((unsigned)(rows & 0xFFFF) << 16));      // dim0 hi | dim1 lo
    g1[3] = (int)(((unsigned)rows >> 16) |
                  ((unsigned)(cols & 0xFFFF) << 16));      // dim1 hi | tile_dim0
    g1[4] = rows & 0xFFFF;                        // tile_dim1 (tile_dim2 = 0)
    g1[5] = row_stride;                           // tensor_dim0_stride lo32
    g1[6] = 0;
    g1[7] = 0;
    i32x4 z4 = {0, 0, 0, 0};
    i32x8 z8 = {0, 0, 0, 0, 0, 0, 0, 0};
    // 6-arg variant (clang-23 / therock headers)
    __builtin_amdgcn_tensor_load_to_lds(g0, g1, z4, z4, z8, 0);
}
#else
#define HAVE_TDM 0
#endif

// ---------- FC GEMM (direct global B): C = relu(A[MxK] * B[KxNc] + bias) ----
// One wave per 16x32 output tile; 4x v_wmma_f32_16x16x32_f16 fully unrolled.
__global__ void k_gemm_fc(const float* __restrict__ A,
                          const float* __restrict__ B,
                          const float* __restrict__ bias,
                          float* __restrict__ C, int M, int Nc) {
    const int wave = threadIdx.x >> 5;
    const int lane = threadIdx.x & 31;
    const int tpr  = Nc >> 5;                           // 16x32 tiles per row
    const int tile = blockIdx.x * (blockDim.x >> 5) + wave;
    if (tile >= (M >> 4) * tpr) return;                 // wave-uniform
    const int mt  = tile / tpr;
    const int cb  = (tile % tpr) << 5;                  // column base
    const int half = lane >> 4;
    const int l15  = lane & 15;
    const int arow = (mt << 4) + l15;

    v8f acc0 = {}, acc1 = {};
#pragma unroll
    for (int k0 = 0; k0 < KDIM; k0 += 32) {
        v16h a = load_a_frag(A + (size_t)arow * KDIM + k0 + half * 16);
        const float* bp = B + (size_t)(k0 + half * 16) * Nc + cb + l15;
        v16h b0, b1;
#pragma unroll
        for (int i = 0; i < 16; ++i) {
            b0[i] = (_Float16)bp[(size_t)i * Nc];
            b1[i] = (_Float16)bp[(size_t)i * Nc + 16];
        }
        acc0 = __builtin_amdgcn_wmma_f32_16x16x32_f16(
            false, a, false, b0, (short)0, acc0, false, false);
        acc1 = __builtin_amdgcn_wmma_f32_16x16x32_f16(
            false, a, false, b1, (short)0, acc1, false, false);
    }
    const int mbase = (mt << 4) + (half << 3);
    const int c0 = cb + l15, c1 = cb + 16 + l15;
    float bv0 = bias[c0], bv1 = bias[c1];
#pragma unroll
    for (int r = 0; r < 8; ++r) {
        C[(size_t)(mbase + r) * Nc + c0] = fmaxf(acc0[r] + bv0, 0.0f);
        C[(size_t)(mbase + r) * Nc + c1] = fmaxf(acc1[r] + bv1, 0.0f);
    }
}

// ---------- Layer GEMM: h = x[Mx64] * W[64xNc], W staged in LDS via TDM -----
__global__ void k_gemm_layer(const float* __restrict__ A,
                             const float* __restrict__ W,
                             float* __restrict__ C, int M, int Nc) {
    __shared__ float sW[KDIM * 64];                 // up to 64x64 f32 = 16 KB
    const int wave = threadIdx.x >> 5;
    const int lane = threadIdx.x & 31;

#if HAVE_TDM
    if (wave == 0) {
        tdm_load_2d_f32((unsigned)(uintptr_t)sW, W, KDIM, Nc, Nc);
        __builtin_amdgcn_s_wait_tensorcnt(0);
    }
#else
    for (int i = threadIdx.x; i < KDIM * Nc; i += blockDim.x) sW[i] = W[i];
#endif
    __syncthreads();

    const int tpr  = Nc >> 5;
    const int tile = blockIdx.x * (blockDim.x >> 5) + wave;
    if (tile >= (M >> 4) * tpr) return;             // wave-uniform
    const int mt  = tile / tpr;
    const int cb  = (tile % tpr) << 5;
    const int half = lane >> 4;
    const int l15  = lane & 15;
    const int arow = (mt << 4) + l15;

    v8f acc0 = {}, acc1 = {};
#pragma unroll
    for (int k0 = 0; k0 < KDIM; k0 += 32) {
        v16h a = load_a_frag(A + (size_t)arow * KDIM + k0 + half * 16);
        const float* bp = sW + (k0 + half * 16) * Nc + cb + l15;   // LDS reads
        v16h b0, b1;
#pragma unroll
        for (int i = 0; i < 16; ++i) {
            b0[i] = (_Float16)bp[i * Nc];
            b1[i] = (_Float16)bp[i * Nc + 16];
        }
        acc0 = __builtin_amdgcn_wmma_f32_16x16x32_f16(
            false, a, false, b0, (short)0, acc0, false, false);
        acc1 = __builtin_amdgcn_wmma_f32_16x16x32_f16(
            false, a, false, b1, (short)0, acc1, false, false);
    }
    const int mbase = (mt << 4) + (half << 3);
#pragma unroll
    for (int r = 0; r < 8; ++r) {
        C[(size_t)(mbase + r) * Nc + cb + l15]      = acc0[r];
        C[(size_t)(mbase + r) * Nc + cb + 16 + l15] = acc1[r];
    }
}

// ---------- attention logits per node ---------------------------------------
__global__ void k_attn(const float* __restrict__ h,
                       const float* __restrict__ a_src,
                       const float* __restrict__ a_dst,
                       float* __restrict__ asrc, float* __restrict__ adst,
                       int N, int dout) {
    int n = blockIdx.x * blockDim.x + threadIdx.x;
    if (n >= N) return;
    const float* hr = h + (size_t)n * dout;
    float s0 = 0.0f, s1 = 0.0f;
    for (int k = 0; k < dout; ++k) {
        float hv = hr[k];
        s0 += hv * a_src[k];
        s1 += hv * a_dst[k];
    }
    asrc[n] = s0;
    adst[n] = s1;
}

// ---------- init softmax state + output accumulator -------------------------
__global__ void k_init(float* __restrict__ out, unsigned* __restrict__ mmax,
                       float* __restrict__ ssum, int N, int dout) {
    int n = blockIdx.x * blockDim.x + threadIdx.x;
    if (n >= N) return;
    mmax[n] = f2mono(__uint_as_float(0xFF800000u));  // -inf
    ssum[n] = 0.0f;
    float* o = out + (size_t)n * dout;
    for (int k = 0; k < dout; ++k) o[k] = 0.0f;
}

// ---------- pass 1: edge logit + segment max --------------------------------
__global__ void k_edge_max(const int* __restrict__ ei, int E, int ETOT,
                           const float* __restrict__ asrc,
                           const float* __restrict__ adst,
                           float* __restrict__ ebuf,
                           unsigned* __restrict__ mmax) {
    int i = blockIdx.x * blockDim.x + threadIdx.x;
    if (i >= ETOT) return;
    int s, d;
    if (i < E) { s = ei[i]; d = ei[E + i]; } else { s = d = i - E; }
    float e = asrc[s] + adst[d];
    e = (e > 0.0f) ? e : NEG_SLOPE * e;              // leaky_relu
    ebuf[i] = e;
    atomicMax(mmax + d, f2mono(e));
}

// ---------- pass 2: exp(e - m[dst]) + segment sum ---------------------------
__global__ void k_edge_exp(const int* __restrict__ ei, int E, int ETOT,
                           const unsigned* __restrict__ mmax,
                           float* __restrict__ ebuf,
                           float* __restrict__ ssum) {
    int i = blockIdx.x * blockDim.x + threadIdx.x;
    if (i >= ETOT) return;
    int d = (i < E) ? ei[E + i] : (i - E);
    float ex = __expf(ebuf[i] - mono2f(mmax[d]));
    ebuf[i] = ex;
    atomicAdd(ssum + d, ex);
}

// ---------- pass 3: out[dst] += coef * h[src] -------------------------------
__global__ void k_scatter(const int* __restrict__ ei, int E, int ETOT,
                          const float* __restrict__ ebuf,
                          const float* __restrict__ ssum,
                          const float* __restrict__ h,
                          float* __restrict__ out, int dout) {
    int t = blockIdx.x * blockDim.x + threadIdx.x;
    int vec = dout >> 2;                              // float4 chunks per edge
    int i = t / vec;
    int q = t - i * vec;
    if (i >= ETOT) return;
    int s, d;
    if (i < E) { s = ei[i]; d = ei[E + i]; } else { s = d = i - E; }
    const float* hr = h + (size_t)s * dout;
    __builtin_prefetch(hr, 0, 0);                     // global_prefetch_b8
    float coef = ebuf[i] / (ssum[d] + 1e-16f);
    float4 hv = ((const float4*)hr)[q];
    float* o = out + (size_t)d * dout + (q << 2);
    atomicAdd(o + 0, coef * hv.x);
    atomicAdd(o + 1, coef * hv.y);
    atomicAdd(o + 2, coef * hv.z);
    atomicAdd(o + 3, coef * hv.w);
}

// ---------- bias (+ optional relu) after aggregation ------------------------
__global__ void k_bias_act(float* __restrict__ out, const float* __restrict__ b,
                           int total, int dout, int do_relu) {
    int t = blockIdx.x * blockDim.x + threadIdx.x;
    if (t >= total) return;
    float v = out[t] + b[t % dout];
    if (do_relu) v = fmaxf(v, 0.0f);
    out[t] = v;
}

extern "C" void kernel_launch(void* const* d_in, const int* in_sizes, int n_in,
                              void* d_out, int out_size, void* d_ws, size_t ws_size,
                              hipStream_t stream) {
    const float* z    = (const float*)d_in[0];
    const int*   ei   = (const int*)d_in[1];
    const float* fc_w = (const float*)d_in[2];
    const float* fc_b = (const float*)d_in[3];
    const float* W[3]    = {(const float*)d_in[4],  (const float*)d_in[8],  (const float*)d_in[12]};
    const float* ASRC[3] = {(const float*)d_in[5],  (const float*)d_in[9],  (const float*)d_in[13]};
    const float* ADST[3] = {(const float*)d_in[6],  (const float*)d_in[10], (const float*)d_in[14]};
    const float* BIAS[3] = {(const float*)d_in[7],  (const float*)d_in[11], (const float*)d_in[15]};

    const int N    = NNODES;
    const int E    = in_sizes[1] / 2;
    const int ETOT = E + N;
    const size_t N64 = (size_t)N * 64;

    float*    ws    = (float*)d_ws;
    float*    xbuf  = ws;                         // N x 64
    float*    pbuf  = ws + N64;                   // N x 64 (ping-pong)
    float*    h     = ws + 2 * N64;               // N x 64
    float*    asrc  = ws + 3 * N64;               // N
    float*    adst  = asrc + N;                   // N
    unsigned* mmax  = (unsigned*)(adst + N);      // N
    float*    ssum  = (float*)mmax + N;           // N
    float*    ebuf  = ssum + N;                   // ETOT

    // FC: x = relu(z @ fc_w + fc_b), [256 x 12544] == reshape [N x 64]
    {
        int M = 256, Nc = 196 * 64;
        int tiles = (M >> 4) * (Nc >> 5);
        k_gemm_fc<<<(tiles + 7) / 8, 256, 0, stream>>>(z, fc_w, fc_b, xbuf, M, Nc);
    }

    float* cur = xbuf;
    float* nxt = pbuf;
    for (int l = 0; l < 3; ++l) {
        const int dout = (l == 2) ? 32 : 64;
        float* dst = (l == 2) ? (float*)d_out : nxt;

        // h = cur @ W_l : WMMA + TDM-staged weights in LDS
        int tiles = (N >> 4) * (dout >> 5);
        k_gemm_layer<<<(tiles + 7) / 8, 256, 0, stream>>>(cur, W[l], h, N, dout);
        k_attn<<<(N + 255) / 256, 256, 0, stream>>>(h, ASRC[l], ADST[l],
                                                    asrc, adst, N, dout);
        k_init<<<(N + 255) / 256, 256, 0, stream>>>(dst, mmax, ssum, N, dout);
        k_edge_max<<<(ETOT + 255) / 256, 256, 0, stream>>>(ei, E, ETOT,
                                                           asrc, adst, ebuf, mmax);
        k_edge_exp<<<(ETOT + 255) / 256, 256, 0, stream>>>(ei, E, ETOT,
                                                           mmax, ebuf, ssum);
        long tsc = (long)ETOT * (dout >> 2);
        k_scatter<<<(int)((tsc + 255) / 256), 256, 0, stream>>>(ei, E, ETOT,
                                                                ebuf, ssum, h,
                                                                dst, dout);
        int total = N * dout;
        k_bias_act<<<(total + 255) / 256, 256, 0, stream>>>(dst, BIAS[l],
                                                            total, dout, l < 2);
        // ping-pong
        float* t2 = cur; cur = dst; nxt = (l == 0) ? t2 : nxt;
        if (l == 1) nxt = t2;
    }
}
